// BiLSTMCRF_75728863363250
// MI455X (gfx1250) — matrix operand order repaired and checked by reading, not compiled, not run on hardware
//
#include <hip/hip_runtime.h>
#include <stdint.h>

typedef __attribute__((ext_vector_type(16))) __bf16 v16bf;
typedef __attribute__((ext_vector_type(8)))  __bf16 v8bf;
typedef __attribute__((ext_vector_type(8)))  float  v8f;
typedef __attribute__((ext_vector_type(4)))  unsigned int v4u;
typedef __attribute__((ext_vector_type(8)))  int    v8i;
typedef __attribute__((ext_vector_type(4)))  int    v4i;

#define NB   64
#define NL   1024
#define NE   256
#define NH   256
#define NG   1024   // 4*H
#define NN2  2048   // both directions
#define NC   6
#define NEGV (-10000.0f)

__device__ __forceinline__ __bf16 f2bf(float f) {
  uint32_t u = __float_as_uint(f);
  uint32_t r = (u + 0x7FFFu + ((u >> 16) & 1u)) >> 16;
  uint16_t s = (uint16_t)r;
  return __builtin_bit_cast(__bf16, s);
}
__device__ __forceinline__ float bf2f(__bf16 h) {
  uint16_t s = __builtin_bit_cast(uint16_t, h);
  return __uint_as_float(((uint32_t)s) << 16);
}
__device__ __forceinline__ float sigf(float x) { return 1.0f / (1.0f + __expf(-x)); }

// A fragment (16x32 bf16, M x K) from a row-major [16][256] bf16 buffer, K-chunk kc.
// Contiguous K order: lanes 0-15 -> M=lane, K=0..15; lanes 16-31 -> K=16..31.
__device__ __forceinline__ v16bf load_a_frag(const __bf16* base, int kc, int lane) {
  int m  = lane & 15;
  int kb = (lane < 16) ? 0 : 16;
  const __bf16* p = base + m * 256 + kc * 32 + kb;
  v8bf lo = *(const v8bf*)(p);
  v8bf hi = *(const v8bf*)(p + 8);
  v16bf a;
#pragma unroll
  for (int i = 0; i < 8; ++i) { a[i] = lo[i]; a[8 + i] = hi[i]; }
  return a;
}

// Async global->LDS copy of 16 bytes per lane (GLOBAL_LOAD_ASYNC_TO_LDS_B128, ASYNCcnt).
__device__ __forceinline__ void async_g2l_b128(uint32_t lds_off, const void* gptr) {
  asm volatile("global_load_async_to_lds_b128 %0, %1, off"
               :: "v"(lds_off), "v"(gptr) : "memory");
}
__device__ __forceinline__ void wait_asynccnt0() {
  asm volatile("s_wait_asynccnt 0x0" ::: "memory");
}

// TDM: one 2D tile (256 rows x 256 bf16) global -> LDS. Issued by one wave.
// 6-arg builtin form: (v4u g0, v8i g1, v4i g2, v4i g3, v8i g4, i32 cpol)
__device__ __forceinline__ void tdm_load_tile_256x256_bf16(uint32_t lds_byte_off,
                                                           const void* gptr) {
  uint64_t ga = (uint64_t)(uintptr_t)gptr;
  v4u g0;
  g0[0] = 1u;                                        // count=1 (valid user descriptor)
  g0[1] = lds_byte_off;                              // lds_addr
  g0[2] = (uint32_t)(ga & 0xFFFFFFFFu);              // global_addr[31:0]
  g0[3] = (uint32_t)((ga >> 32) & 0x01FFFFFFu)       // global_addr[56:32]
          | (2u << 30);                              // type=2 ("image")
  v8i g1;
  g1[0] = (int)(1u << 16);        // workgroup_mask=0, data_size=1 (2 bytes)
  g1[1] = (int)(256u << 16);      // tensor_dim0[15:0]=256 in bits[31:16]
  g1[2] = (int)(256u << 16);      // tensor_dim0 hi=0 | tensor_dim1[15:0]=256
  g1[3] = (int)(256u << 16);      // tensor_dim1 hi=0 | tile_dim0=256
  g1[4] = (int)(256u);            // tile_dim1=256, tile_dim2=0
  g1[5] = (int)(256);             // tensor_dim0_stride[31:0]=256
  g1[6] = 0;                      // stride0 hi, stride1 lo
  g1[7] = 0;
  v4i gz4 = {0, 0, 0, 0};
  v8i gz8 = {0, 0, 0, 0, 0, 0, 0, 0};
  __builtin_amdgcn_tensor_load_to_lds(g0, g1, gz4, gz4, gz8, 0);
}

// ---------------- Kernel 1: weight conversion / permutation + zero-init ----------------
// Whhp layout: [dir][slice(4)][gate(4)][jl(64)][k(256)]  (per-WG contiguous 128KB blocks)
__global__ void prep_weights(const float* __restrict__ Wih_f, const float* __restrict__ Wih_b,
                             const float* __restrict__ Whh_f, const float* __restrict__ Whh_b,
                             __bf16* __restrict__ Wihc, __bf16* __restrict__ Whhp,
                             __bf16* __restrict__ hx, int* __restrict__ cnt) {
  int i = blockIdx.x * blockDim.x + threadIdx.x;
  if (i < NN2 * NE) {
    int n = i >> 8, k = i & 255;
    float v = (n < NG) ? Wih_f[n * NE + k] : Wih_b[(n - NG) * NE + k];
    Wihc[i] = f2bf(v);
  }
  if (i < 2 * NG * NH) {
    int dir = i / (NG * NH);
    int r = i % (NG * NH);
    int n = r >> 8, k = r & 255;           // n = g*256 + j
    int g = n >> 8, j = n & 255;
    int slice = j >> 6, jl = j & 63;
    const float* W = dir ? Whh_b : Whh_f;
    size_t out = ((((size_t)dir * 4 + slice) * 4 + g) * 64 + jl) * 256 + k;
    Whhp[out] = f2bf(W[n * NH + k]);
  }
  if (i < 2 * 2 * NB * NH) hx[i] = f2bf(0.0f);   // zero both parities, both dirs
  if (i < 2 * NL) cnt[i] = 0;                    // per-step barrier counters
}

// -------- Kernel 2: fused embedding gather + input-projection GEMM (WMMA) --------
__global__ __launch_bounds__(256) void gemm_xw(
    const int* __restrict__ sentences, const float* __restrict__ embed,
    const __bf16* __restrict__ Wihc, const float* __restrict__ b_f,
    const float* __restrict__ b_b, __bf16* __restrict__ xw) {
  __shared__ __align__(32) __bf16 Alds[16 * 256];
  int tid  = threadIdx.x;
  int row0 = blockIdx.x * 16;
  int r  = tid >> 4;
  int e0 = (tid & 15) * 16;
  int tok = sentences[row0 + r];
  const float* ep = embed + (size_t)tok * NE + e0;
#pragma unroll
  for (int e = 0; e < 16; ++e) Alds[r * 256 + e0 + e] = f2bf(ep[e]);
  __syncthreads();

  int wave = tid >> 5, lane = tid & 31;
  int nl = lane & 15;
  int k0 = (lane < 16) ? 0 : 16;
  int mb = (lane < 16) ? 0 : 8;
#pragma unroll 1
  for (int i = 0; i < 16; ++i) {
    int nt = wave * 16 + i;
    int n2 = nt * 16 + nl;
    float bias = (n2 < NG) ? b_f[n2] : b_b[n2 - NG];
    if (i < 15) __builtin_prefetch(Wihc + (size_t)(n2 + 16) * NE, 0, 1);
    v8f acc;
#pragma unroll
    for (int q = 0; q < 8; ++q) acc[q] = bias;
#pragma unroll
    for (int kc = 0; kc < 8; ++kc) {
      v16bf a = load_a_frag(Alds, kc, lane);
      v16bf b = *(const v16bf*)(Wihc + (size_t)n2 * NE + kc * 32 + k0);
      acc = __builtin_amdgcn_wmma_f32_16x16x32_bf16(false, a, false, b,
                                                    (short)0, acc, false, false);
    }
#pragma unroll
    for (int q = 0; q < 8; ++q) {
      int m = q + mb;
      xw[(size_t)(row0 + m) * NN2 + n2] = f2bf(acc[q]);
    }
  }
}

// ---------- Kernel 3: recurrent LSTM. 8 WGs = 2 dirs x 4 hidden slices. ----------
// Each WG: all 64 batch rows, hidden units [slice*64, slice*64+64), all 4 gates.
// W slice resident in LDS (TDM-loaded once); h exchanged per step via global
// double buffer + per-step counter barrier; h staged to LDS with async b128 loads.
__global__ __launch_bounds__(512) void lstm_step(
    const __bf16* __restrict__ xw, const __bf16* __restrict__ Whhp,
    const int* __restrict__ masks, __bf16* __restrict__ hbuf,
    __bf16* __restrict__ hx, int* __restrict__ cnt) {
  extern __shared__ char smem[];
  __bf16* Wlds  = (__bf16*)smem;                       // 4*64*256 bf16 = 128KB
  __bf16* h_lds = (__bf16*)(smem + 131072);            // 64*256 bf16  =  32KB
  float*  c_lds = (float*)(smem + 131072 + 32768);     // 64*64 fp32   =  16KB

  int dir = blockIdx.x >> 2;
  int slice = blockIdx.x & 3;
  int js0 = slice * 64;
  int tid = threadIdx.x, wave = tid >> 5, lane = tid & 31;
  int mt = wave & 3;          // batch 16-row tile
  int q  = wave >> 2;         // jl 16-tile
  int nl = lane & 15;
  int k0 = (lane < 16) ? 0 : 16;
  int mb = (lane < 16) ? 0 : 8;

  for (int i = tid; i < 64 * 64; i += 512) c_lds[i] = 0.0f;

  // one-time TDM load of this WG's W slice (256 rows x 256 bf16) into LDS
  if (tid < 32) {
    const __bf16* wsrc = Whhp + ((size_t)(dir * 4 + slice)) * (256 * 256);
    tdm_load_tile_256x256_bf16((uint32_t)(uintptr_t)Wlds, wsrc);
    __builtin_amdgcn_s_wait_tensorcnt(0);
  }
  __syncthreads();

  int* mycnt = cnt + dir * NL;

#pragma unroll 1
  for (int t = 0; t < NL; ++t) {
    int pos = dir ? (NL - 1 - t) : t;
    int p = t & 1;
    // stage full h(prev) [64][256] bf16 from global into LDS (async path, 32KB)
    {
      const __bf16* hsrc = hx + ((size_t)(dir * 2 + p) * NB) * NH;
      uint32_t hbase = (uint32_t)(uintptr_t)h_lds;
#pragma unroll
      for (int i = 0; i < 4; ++i) {
        int idx = i * 512 + tid;               // 16-byte chunk index (2048 total)
        async_g2l_b128(hbase + idx * 16, (const void*)(hsrc + idx * 8));
      }
      wait_asynccnt0();
    }
    __syncthreads();

    // gates for (batch tile mt, jl tile q, all 4 gates); init acc from xw
    v8f acc[4];
#pragma unroll
    for (int g = 0; g < 4; ++g) {
      int n2 = dir * NG + g * NH + js0 + q * 16 + nl;
#pragma unroll
      for (int r = 0; r < 8; ++r) {
        int bg = mt * 16 + r + mb;
        acc[g][r] = bf2f(xw[((size_t)bg * NL + pos) * NN2 + n2]);
      }
    }
#pragma unroll
    for (int kc = 0; kc < 8; ++kc) {
      v16bf a = load_a_frag(h_lds + (size_t)mt * 16 * 256, kc, lane);
#pragma unroll
      for (int g = 0; g < 4; ++g) {
        int nloc = g * 64 + q * 16 + nl;
        v16bf b = *(const v16bf*)(Wlds + (size_t)nloc * 256 + kc * 32 + k0);
        acc[g] = __builtin_amdgcn_wmma_f32_16x16x32_bf16(false, a, false, b,
                                                         (short)0, acc[g],
                                                         false, false);
      }
    }

    // gate fusion + state update (wave-local patch: 16 batch x 16 jl)
    __bf16* hdst = hx + ((size_t)(dir * 2 + (1 - p)) * NB) * NH;
#pragma unroll
    for (int r = 0; r < 8; ++r) {
      int bg = mt * 16 + r + mb;
      int jl = q * 16 + nl;
      int j  = js0 + jl;
      float iv = sigf(acc[0][r]);
      float fv = sigf(acc[1][r]);
      float gv = tanhf(acc[2][r]);
      float ov = sigf(acc[3][r]);
      float c_old = c_lds[bg * 64 + jl];
      float c_new = fv * c_old + iv * gv;
      float h_new = ov * tanhf(c_new);
      bool mk = masks[bg * NL + pos] != 0;
      float h_old = bf2f(h_lds[bg * 256 + j]);
      c_lds[bg * 64 + jl] = mk ? c_new : c_old;
      hdst[(size_t)bg * NH + j] = f2bf(mk ? h_new : h_old);
      hbuf[(((size_t)dir * NB + bg) * NL + pos) * NH + j] = f2bf(mk ? h_new : 0.0f);
    }

    // inter-WG barrier: all 4 slices of this direction finished writing h(next)
    if (t + 1 < NL) {
      __syncthreads();
      if (tid == 0) {
        __threadfence();
        atomicAdd(&mycnt[t], 1);
        while (__hip_atomic_load(&mycnt[t], __ATOMIC_ACQUIRE,
                                 __HIP_MEMORY_SCOPE_AGENT) < 4) {
          __builtin_amdgcn_s_sleep(1);
        }
        __threadfence();
      }
      __syncthreads();
    }
  }
}

// ---------------- Kernel 4: output projection (C=6, tiny) ----------------
__global__ void feats_kernel(const __bf16* __restrict__ hbuf, const float* __restrict__ W_out,
                             const float* __restrict__ b_out, float* __restrict__ feats) {
  int idx = blockIdx.x * blockDim.x + threadIdx.x;   // b*L + l
  if (idx >= NB * NL) return;
  const __bf16* hf = hbuf + (size_t)idx * NH;
  const __bf16* hb = hbuf + ((size_t)NB * NL + idx) * NH;
  float acc[NC];
#pragma unroll
  for (int c = 0; c < NC; ++c) acc[c] = b_out[c];
  for (int k = 0; k < NH; ++k) {
    float hv = bf2f(hf[k]);
#pragma unroll
    for (int c = 0; c < NC; ++c) acc[c] += hv * W_out[c * (2 * NH) + k];
  }
  for (int k = 0; k < NH; ++k) {
    float hv = bf2f(hb[k]);
#pragma unroll
    for (int c = 0; c < NC; ++c) acc[c] += hv * W_out[c * (2 * NH) + NH + k];
  }
#pragma unroll
  for (int c = 0; c < NC; ++c) feats[(size_t)idx * NC + c] = acc[c];
}

// ---------------- Kernel 5: Viterbi decode, one thread per batch row ----------------
__global__ void viterbi_kernel(const float* __restrict__ feats, const int* __restrict__ masks,
                               const float* __restrict__ trans, unsigned char* __restrict__ bps,
                               float* __restrict__ out) {
  int b = threadIdx.x;
  if (b >= NB) return;
  float tr[NC * NC];
#pragma unroll
  for (int i = 0; i < NC * NC; ++i) tr[i] = trans[i];
  float score[NC];
#pragma unroll
  for (int i = 0; i < NC; ++i) score[i] = NEGV;
  score[4] = 0.0f;                       // START
  for (int t = 0; t < NL; ++t) {
    bool mk = masks[b * NL + t] != 0;
    float ns[NC];
#pragma unroll
    for (int i = 0; i < NC; ++i) {
      float best = score[0] + tr[i * NC + 0]; int bj = 0;
#pragma unroll
      for (int j = 1; j < NC; ++j) {
        float v = score[j] + tr[i * NC + j];
        if (v > best) { best = v; bj = j; }
      }
      ns[i] = best + feats[((size_t)b * NL + t) * NC + i];
      bps[((size_t)t * NB + b) * NC + i] = (unsigned char)bj;
    }
#pragma unroll
    for (int i = 0; i < NC; ++i) score[i] = mk ? ns[i] : score[i];
  }
#pragma unroll
  for (int i = 0; i < NC; ++i) score[i] += tr[5 * NC + i];
  float best = score[0]; int tag = 0;
#pragma unroll
  for (int i = 1; i < NC; ++i) if (score[i] > best) { best = score[i]; tag = i; }
  out[b] = best;
  float* paths = out + NB;
  paths[(size_t)b * NL + (NL - 1)] = (float)tag;
  for (int t = NL - 1; t >= 1; --t) {
    bool mk = masks[b * NL + t] != 0;
    tag = mk ? (int)bps[((size_t)t * NB + b) * NC + tag] : tag;
    paths[(size_t)b * NL + (t - 1)] = (float)tag;
  }
}

extern "C" void kernel_launch(void* const* d_in, const int* in_sizes, int n_in,
                              void* d_out, int out_size, void* d_ws, size_t ws_size,
                              hipStream_t stream) {
  const int*   sentences = (const int*)d_in[0];
  const int*   masks     = (const int*)d_in[1];
  const float* embed     = (const float*)d_in[2];
  const float* W_ih_f    = (const float*)d_in[3];
  const float* W_hh_f    = (const float*)d_in[4];
  const float* b_f       = (const float*)d_in[5];
  const float* W_ih_b    = (const float*)d_in[6];
  const float* W_hh_b    = (const float*)d_in[7];
  const float* b_b       = (const float*)d_in[8];
  const float* W_out     = (const float*)d_in[9];
  const float* b_out     = (const float*)d_in[10];
  const float* trans     = (const float*)d_in[11];
  float* out = (float*)d_out;

  char* ws = (char*)d_ws;
  __bf16* Wihc = (__bf16*)ws;  ws += (size_t)NN2 * NE * 2;          //   1 MB
  __bf16* Whhp = (__bf16*)ws;  ws += (size_t)2 * NG * NH * 2;       //   1 MB
  __bf16* xw   = (__bf16*)ws;  ws += (size_t)NB * NL * NN2 * 2;     // 256 MB
  __bf16* hbuf = (__bf16*)ws;  ws += (size_t)2 * NB * NL * NH * 2;  //  64 MB
  float* feats = (float*)ws;   ws += (size_t)NB * NL * NC * 4;      // 1.5 MB
  unsigned char* bps = (unsigned char*)ws; ws += (size_t)NL * NB * NC; // 384 KB
  __bf16* hx = (__bf16*)ws;    ws += (size_t)2 * 2 * NB * NH * 2;   // 128 KB
  int* cnt = (int*)ws;                                              //   8 KB

  prep_weights<<<(NN2 * NE + 255) / 256, 256, 0, stream>>>(
      W_ih_f, W_ih_b, W_hh_f, W_hh_b, Wihc, Whhp, hx, cnt);
  gemm_xw<<<(NB * NL) / 16, 256, 0, stream>>>(
      sentences, embed, Wihc, b_f, b_b, xw);
  lstm_step<<<8, 512, 176 * 1024, stream>>>(xw, Whhp, masks, hbuf, hx, cnt);
  feats_kernel<<<(NB * NL + 255) / 256, 256, 0, stream>>>(hbuf, W_out, b_out, feats);
  viterbi_kernel<<<1, 64, 0, stream>>>(feats, masks, trans, bps, out);
}
